// XLNetRelativeAttention_523986010653
// MI455X (gfx1250) — matrix operand order repaired and checked by reading, not compiled, not run on hardware
//
#include <hip/hip_runtime.h>
#include <hip/hip_bf16.h>

typedef __attribute__((ext_vector_type(16))) _Float16     v16h;
typedef __attribute__((ext_vector_type(8)))  _Float16     v8h;
typedef __attribute__((ext_vector_type(2)))  _Float16     v2h;
typedef __attribute__((ext_vector_type(2)))  __fp16       fp16x2;
typedef __attribute__((ext_vector_type(8)))  float        v8f;
typedef __attribute__((ext_vector_type(4)))  float        f32x4;
typedef __attribute__((ext_vector_type(2)))  float        f32x2;
typedef __attribute__((ext_vector_type(4)))  unsigned int u32x4;
typedef __attribute__((ext_vector_type(4)))  int          v4i;

union HFrag { u32x4 u[2]; v16h v; };

#define QLEN 512
#define BSZ  8
#define NH   16
#define DH   64
#define DM   1024
#define RLEN 1024

#if defined(__gfx1250__) && __has_builtin(__builtin_amdgcn_global_load_async_to_lds_b128) && __has_builtin(__builtin_amdgcn_s_wait_asynccnt)
#define USE_ASYNC_LDS 1
#else
#define USE_ASYNC_LDS 0
#endif

__device__ __forceinline__ v2h pk(float a, float b) {
  union { fp16x2 i; v2h o; } u;
  u.i = __builtin_amdgcn_cvt_pkrtz(a, b);
  return u.o;
}

// ---------------------------------------------------------------------------
// Generic GEMM: C[M,1024] = A[M,1024] (f32) x W[1024,1024] (f32), f16 WMMA.
// MODE 0: out f32 scattered to [b][n][row][d]   (Q heads, biases added later)
// MODE 1: out f16 scattered to [b][n][row][d]   (K/V/Kr heads)
// MODE 2: out f32 row-major [m][1024] + residual (output projection)
// WT: read W as [n][k] (for wo^T) instead of [k][n].
// ---------------------------------------------------------------------------
template <int MODE, int WT>
__global__ __launch_bounds__(128) void proj_gemm_f16(
    const float* __restrict__ A, const float* __restrict__ W,
    int L, const float* __restrict__ residual,
    float* __restrict__ outF, _Float16* __restrict__ outH)
{
  __shared__ __align__(16) _Float16 As[64 * 32];   // [m][k]
  __shared__ __align__(16) _Float16 Bs[64 * 32];   // [n][k]  (transposed)
  const int tid  = threadIdx.x;
  const int lane = tid & 31;
  const int wid  = tid >> 5;
  const int half = lane >> 4;
  const int l15  = lane & 15;
  const long mBase = (long)blockIdx.y * 64;
  const int  nBase = blockIdx.x * 64;

  v8f acc[4] = {};

  for (int k0 = 0; k0 < DM; k0 += 32) {
    // --- stage A tile 64x32, f32 -> f16, vectorized ---
    {
      const int row = tid >> 1;
      const int kp  = (tid & 1) * 16;
      const f32x4* src = (const f32x4*)(A + (mBase + row) * DM + k0 + kp);
      f32x4 f0 = src[0], f1 = src[1], f2 = src[2], f3 = src[3];
      v2h* dst = (v2h*)&As[row * 32 + kp];
      dst[0] = pk(f0.x, f0.y); dst[1] = pk(f0.z, f0.w);
      dst[2] = pk(f1.x, f1.y); dst[3] = pk(f1.z, f1.w);
      dst[4] = pk(f2.x, f2.y); dst[5] = pk(f2.z, f2.w);
      dst[6] = pk(f3.x, f3.y); dst[7] = pk(f3.z, f3.w);
    }
    // --- stage W tile transposed into Bs[n][k], vectorized global reads ---
    if (WT == 0) {
#pragma unroll
      for (int e = 0; e < 4; ++e) {
        const int quad = tid + e * 128;      // 512 quads = 32k x 16 ngroups
        const int kk = quad & 31;
        const int ng = quad >> 5;
        const f32x4 w = *(const f32x4*)(W + (long)(k0 + kk) * DM + nBase + ng * 4);
        _Float16* dcol = &Bs[(ng * 4) * 32 + kk];
        dcol[0]  = (_Float16)w.x;
        dcol[32] = (_Float16)w.y;
        dcol[64] = (_Float16)w.z;
        dcol[96] = (_Float16)w.w;
      }
    } else {
#pragma unroll
      for (int e = 0; e < 4; ++e) {
        const int quad = tid + e * 128;      // 512 quads = 64n x 8 kgroups
        const int nn = quad & 63;
        const int kg = quad >> 6;
        const f32x4 w = *(const f32x4*)(W + (long)(nBase + nn) * DM + k0 + kg * 4);
        v2h* d = (v2h*)&Bs[nn * 32 + kg * 4];
        d[0] = pk(w.x, w.y);
        d[1] = pk(w.z, w.w);
      }
    }
    __syncthreads();

    HFrag a;
    {
      const _Float16* ap = &As[(wid * 16 + l15) * 32];
      a.u[0] = *(const u32x4*)(ap + half * 8);
      a.u[1] = *(const u32x4*)(ap + 16 + half * 8);
    }
#pragma unroll
    for (int nt = 0; nt < 4; ++nt) {
      HFrag b;
      const u32x4* bp = (const u32x4*)&Bs[(nt * 16 + l15) * 32 + half * 8];
      b.u[0] = bp[0]; b.u[1] = bp[2];
      acc[nt] = __builtin_amdgcn_wmma_f32_16x16x32_f16(
          false, a.v, false, b.v, (short)0, acc[nt], false, false);
    }
    __syncthreads();
  }

#pragma unroll
  for (int nt = 0; nt < 4; ++nt) {
#pragma unroll
    for (int r = 0; r < 8; ++r) {
      const long gm  = mBase + wid * 16 + r + 8 * half;
      const int  gn  = nBase + nt * 16 + l15;
      const float v  = acc[nt][r];
      if (MODE == 2) {
        const long off = gm * DM + gn;
        outF[off] = v + residual[off];
      } else {
        const int  bb = (int)(gm & 7);          // m = i*BSZ + b
        const long ii = gm >> 3;
        const int  hn = gn >> 6;                // gn = n*64 + d
        const int  dd = gn & 63;
        const long off = (((long)(bb * NH + hn)) * L + ii) * DH + dd;
        if (MODE == 0) outF[off] = v;
        else           outH[off] = (_Float16)v;
      }
    }
  }
}

// ---------------------------------------------------------------------------
// Attention core for one (b, n, 16-row i-tile). 4 wave32 per block.
// ac via WMMA, bd via WMMA on aligned jr tiles + analytic rel-shift
// (bd[i,j] = Qr[i] . Kr[j + 512 - i]), ef/mask/scale + softmax on VALU,
// P·V via WMMA with probabilities regenerated into A-fragment layout.
// ---------------------------------------------------------------------------
__global__ __launch_bounds__(128) void attn_core_f16(
    const float* __restrict__ q,        // [b][n][QLEN][DH] f32
    const _Float16* __restrict__ kh,    // [b][n][QLEN][DH]
    const _Float16* __restrict__ vh,    // [b][n][QLEN][DH]
    const _Float16* __restrict__ kr,    // [b][n][RLEN][DH]
    const float* __restrict__ maskF,    // [i][j][b][nh]
    const float* __restrict__ segmat,   // [i][j][b][2]
    const float* __restrict__ rw,
    const float* __restrict__ rr,
    const float* __restrict__ rs,
    const float* __restrict__ segemb,   // [2][nh][DH]
    float* __restrict__ outv)           // [(i*BSZ+b)][DM]
{
  __shared__ float S[16 * 512];
  __shared__ __align__(16) _Float16 Qw[16 * 64];
  __shared__ __align__(16) _Float16 Qr[16 * 64];
  __shared__ __align__(16) _Float16 Vt[64 * 32];  // [d][j] transposed V chunk
  __shared__ float efs[16 * 2];
  __shared__ float redbuf[128];
  __shared__ float rowmax[16];
  __shared__ float rowrcp[16];

  const int tid  = threadIdx.x;
  const int lane = tid & 31;
  const int wid  = tid >> 5;
  const int half = lane >> 4;
  const int l15  = lane & 15;
  const int i0 = blockIdx.x * 16;
  const int n  = blockIdx.y;
  const int b  = blockIdx.z;
  const long bn = (long)b * NH + n;

  // stage (q + r_w_bias) and (q + r_r_bias) as f16, vectorized
  {
    const int row = tid >> 3;
    const int dp  = (tid & 7) * 8;
    const f32x4* src = (const f32x4*)(q + (bn * QLEN + i0 + row) * DH + dp);
    const f32x4 q0 = src[0], q1 = src[1];
    const f32x4 w0 = *(const f32x4*)(rw + n * DH + dp);
    const f32x4 w1 = *(const f32x4*)(rw + n * DH + dp + 4);
    const f32x4 r0 = *(const f32x4*)(rr + n * DH + dp);
    const f32x4 r1 = *(const f32x4*)(rr + n * DH + dp + 4);
    v2h* dW = (v2h*)&Qw[row * 64 + dp];
    v2h* dR = (v2h*)&Qr[row * 64 + dp];
    dW[0] = pk(q0.x + w0.x, q0.y + w0.y); dW[1] = pk(q0.z + w0.z, q0.w + w0.w);
    dW[2] = pk(q1.x + w1.x, q1.y + w1.y); dW[3] = pk(q1.z + w1.z, q1.w + w1.w);
    dR[0] = pk(q0.x + r0.x, q0.y + r0.y); dR[1] = pk(q0.z + r0.z, q0.w + r0.w);
    dR[2] = pk(q1.x + r1.x, q1.y + r1.y); dR[3] = pk(q1.z + r1.z, q1.w + r1.w);
  }
  // ef[row][s] = sum_d (q + r_s_bias) * seg_embed[s,n,:]
  if (tid < 32) {
    const int row = tid >> 1;
    const int s   = tid & 1;
    const float* src = q + (bn * QLEN + i0 + row) * DH;
    const float* se  = segemb + (s * NH + n) * DH;
    float a = 0.f;
    for (int d = 0; d < DH; ++d) a += (src[d] + rs[n * DH + d]) * se[d];
    efs[row * 2 + s] = a;
  }
  __syncthreads();

  // ---- ac: each wave owns j-chunks {wid, wid+4} of eight 64-wide chunks ----
  for (int jc = wid; jc < 8; jc += 4) {
    const int j0 = jc * 64;
    v8f acc[4] = {};
#pragma unroll
    for (int ks = 0; ks < 2; ++ks) {
      const int k0 = ks * 32;
      HFrag a;
      const _Float16* ap = &Qw[l15 * 64 + k0];
      a.u[0] = *(const u32x4*)(ap + half * 8);
      a.u[1] = *(const u32x4*)(ap + 16 + half * 8);
#pragma unroll
      for (int nt = 0; nt < 4; ++nt) {
        HFrag bf;
        const _Float16* bp = kh + (bn * QLEN + j0 + nt * 16 + l15) * DH + k0 + half * 8;
        bf.u[0] = ((const u32x4*)bp)[0];
        bf.u[1] = ((const u32x4*)bp)[2];
        acc[nt] = __builtin_amdgcn_wmma_f32_16x16x32_f16(
            false, a.v, false, bf.v, (short)0, acc[nt], false, false);
      }
    }
#pragma unroll
    for (int nt = 0; nt < 4; ++nt)
#pragma unroll
      for (int r = 0; r < 8; ++r)
        S[(r + 8 * half) * 512 + j0 + nt * 16 + l15] = acc[nt][r];
  }
  __syncthreads();   // S fully written before bd accumulates into it

  // ---- bd over the aligned jr band, tiles round-robin across waves ----
  {
    const int jr_lo = QLEN - i0 - 15;      // >= 1
    const int jr_hi = RLEN - 1 - i0;       // <= 1023
    int c = 0;
    for (int jt = (jr_lo / 64) * 64; jt <= jr_hi; jt += 64, ++c) {
      if ((c & 3) != wid) continue;
      v8f acc[4] = {};
#pragma unroll
      for (int ks = 0; ks < 2; ++ks) {
        const int k0 = ks * 32;
        HFrag a;
        const _Float16* ap = &Qr[l15 * 64 + k0];
        a.u[0] = *(const u32x4*)(ap + half * 8);
        a.u[1] = *(const u32x4*)(ap + 16 + half * 8);
#pragma unroll
        for (int nt = 0; nt < 4; ++nt) {
          HFrag bf;
          const _Float16* bp = kr + (bn * RLEN + jt + nt * 16 + l15) * DH + k0 + half * 8;
          bf.u[0] = ((const u32x4*)bp)[0];
          bf.u[1] = ((const u32x4*)bp)[2];
          acc[nt] = __builtin_amdgcn_wmma_f32_16x16x32_f16(
              false, a.v, false, bf.v, (short)0, acc[nt], false, false);
        }
      }
#pragma unroll
      for (int nt = 0; nt < 4; ++nt)
#pragma unroll
        for (int r = 0; r < 8; ++r) {
          const int ml = r + 8 * half;
          const int ig = i0 + ml;
          const int jr = jt + nt * 16 + l15;
          const int jcol = jr - QLEN + ig;        // rel_shift
          if ((unsigned)jcol < (unsigned)QLEN) S[ml * 512 + jcol] += acc[nt][r];
        }
    }
  }
  __syncthreads();

  // ---- ef + mask + scale (dominant HBM traffic: prefetch ahead) ----
#pragma unroll 4
  for (int e = 0; e < 64; ++e) {
    const int idx = tid + e * 128;           // 16*512 elements
    const int ml = idx >> 9;
    const int j  = idx & 511;
    const long ig = i0 + ml;
    const long ij = (ig * QLEN + j) * BSZ + b;
    if (e + 8 < 64) {
      const int idx2 = tid + (e + 8) * 128;
      const long ij2 = (((long)(i0 + (idx2 >> 9))) * QLEN + (idx2 & 511)) * BSZ + b;
      __builtin_prefetch(segmat + ij2 * 2, 0, 1);
      __builtin_prefetch(maskF + ij2 * NH, 0, 1);
    }
    const f32x2 sm = *(const f32x2*)(segmat + ij * 2);
    const float mv = maskF[ij * NH + n];
    float v = S[ml * 512 + j] + sm.x * efs[ml * 2] + sm.y * efs[ml * 2 + 1];
    S[ml * 512 + j] = v * 0.125f - 1e30f * mv;
  }
  __syncthreads();

  // ---- softmax row stats: 8 threads per row ----
  {
    const int srow = tid >> 3;
    const int seg  = tid & 7;
    float m = -3.4e38f;
    for (int j = seg * 64; j < seg * 64 + 64; ++j) m = fmaxf(m, S[srow * 512 + j]);
    redbuf[tid] = m;
    __syncthreads();
    if (seg == 0) {
      float mm = redbuf[tid];
#pragma unroll
      for (int t = 1; t < 8; ++t) mm = fmaxf(mm, redbuf[tid + t]);
      rowmax[srow] = mm;
    }
    __syncthreads();
    const float mm = rowmax[srow];
    float s = 0.f;
    for (int j = seg * 64; j < seg * 64 + 64; ++j) s += __expf(S[srow * 512 + j] - mm);
    redbuf[tid] = s;
    __syncthreads();
    if (seg == 0) {
      float ss = redbuf[tid];
#pragma unroll
      for (int t = 1; t < 8; ++t) ss += redbuf[tid + t];
      rowrcp[srow] = 1.f / ss;
    }
    __syncthreads();
  }

  // ---- P · V : wave wid owns d-subtile nt = wid ----
  v8f o = {};
  for (int ks = 0; ks < 16; ++ks) {
    const int j0 = ks * 32;
#pragma unroll
    for (int e = 0; e < 2; ++e) {          // stage V^T chunk [64d][32j], vector reads
      const int c = tid + e * 128;         // 256 chunks = 32j x 8 dgroups
      const int j  = c & 31;
      const int dg = c >> 5;
      const v8h vv = *(const v8h*)(vh + (bn * QLEN + j0 + j) * DH + dg * 8);
#pragma unroll
      for (int t = 0; t < 8; ++t) Vt[(dg * 8 + t) * 32 + j] = vv[t];
    }
    __syncthreads();
    const int ml = l15;
    const float rmax = rowmax[ml];
    const float rcp  = rowrcp[ml];
    v16h av;
#pragma unroll
    for (int e2 = 0; e2 < 16; ++e2) {      // A-fragment k layout (16-bit A 16x32)
      const int vv = e2 >> 1, p = e2 & 1;
      const int jj = (vv < 4) ? (half * 8 + vv * 2 + p)
                              : (16 + half * 8 + (vv - 4) * 2 + p);
      const float sv = S[ml * 512 + j0 + jj];
      av[e2] = (_Float16)(__expf(sv - rmax) * rcp);
    }
    HFrag bf;
    const u32x4* bp = (const u32x4*)&Vt[(wid * 16 + l15) * 32 + half * 8];
    bf.u[0] = bp[0]; bf.u[1] = bp[2];
    o = __builtin_amdgcn_wmma_f32_16x16x32_f16(
        false, av, false, bf.v, (short)0, o, false, false);
    __syncthreads();
  }
#pragma unroll
  for (int r = 0; r < 8; ++r) {
    const long ig = i0 + r + 8 * half;
    const int d = wid * 16 + l15;
    outv[(ig * BSZ + b) * DM + n * DH + d] = o[r];
  }
}

// ---------------------------------------------------------------------------
// LayerNorm over last dim (1024), one block per row.
// Row staged to LDS with gfx1250 async global->LDS copy when available.
// ---------------------------------------------------------------------------
__global__ __launch_bounds__(256) void layernorm_k(
    const float* __restrict__ x, const float* __restrict__ gamma,
    const float* __restrict__ beta, float* __restrict__ out)
{
  __shared__ float red[256];
  __shared__ __align__(16) float xs[DM];
  const long row = blockIdx.x;
  const int tid = threadIdx.x;
  const float* xr = x + row * DM;

#if USE_ASYNC_LDS
  // each of 256 lanes copies 16 bytes: one full 4KB row per instruction issue
  __builtin_amdgcn_global_load_async_to_lds_b128(
      (__attribute__((address_space(1))) v4i*)(xr + tid * 4),
      (__attribute__((address_space(3))) v4i*)&xs[tid * 4], 0, 0);
  __builtin_amdgcn_s_wait_asynccnt(0);
  __syncthreads();
#else
  *(f32x4*)&xs[tid * 4] = *(const f32x4*)(xr + tid * 4);
  __syncthreads();
#endif

  float lv[4];
  float s = 0.f;
#pragma unroll
  for (int e = 0; e < 4; ++e) { lv[e] = xs[tid + e * 256]; s += lv[e]; }
  red[tid] = s; __syncthreads();
  for (int off = 128; off > 0; off >>= 1) {
    if (tid < off) red[tid] += red[tid + off];
    __syncthreads();
  }
  const float mu = red[0] * (1.f / DM);
  __syncthreads();
  float vs = 0.f;
#pragma unroll
  for (int e = 0; e < 4; ++e) { const float d = lv[e] - mu; vs += d * d; }
  red[tid] = vs; __syncthreads();
  for (int off = 128; off > 0; off >>= 1) {
    if (tid < off) red[tid] += red[tid + off];
    __syncthreads();
  }
  const float rstd = rsqrtf(red[0] * (1.f / DM) + 1e-12f);
  float* orow = out + row * DM;
#pragma unroll
  for (int e = 0; e < 4; ++e) {
    const int c = tid + e * 256;
    orow[c] = (lv[e] - mu) * rstd * gamma[c] + beta[c];
  }
}

// ---------------------------------------------------------------------------
extern "C" void kernel_launch(void* const* d_in, const int* in_sizes, int n_in,
                              void* d_out, int out_size, void* d_ws, size_t ws_size,
                              hipStream_t stream) {
  const float* h      = (const float*)d_in[0];
  const float* g      = (const float*)d_in[1];
  const float* r      = (const float*)d_in[2];
  const float* mask_h = (const float*)d_in[3];
  const float* mask_g = (const float*)d_in[4];
  const float* segmat = (const float*)d_in[5];
  const float* wq     = (const float*)d_in[6];
  const float* wk     = (const float*)d_in[7];
  const float* wv     = (const float*)d_in[8];
  const float* wo     = (const float*)d_in[9];
  const float* wr     = (const float*)d_in[10];
  const float* rwb    = (const float*)d_in[11];
  const float* rrb    = (const float*)d_in[12];
  const float* rsb    = (const float*)d_in[13];
  const float* segemb = (const float*)d_in[14];
  const float* gamma  = (const float*)d_in[15];
  const float* beta   = (const float*)d_in[16];

  char* ws = (char*)d_ws;
  size_t off = 0;
  auto alloc = [&](size_t bytes) { char* p = ws + off; off += bytes; return p; };
  float*    qh  = (float*)   alloc((size_t)BSZ * NH * QLEN * DH * 4);
  float*    qg  = (float*)   alloc((size_t)BSZ * NH * QLEN * DH * 4);
  _Float16* kbf = (_Float16*)alloc((size_t)BSZ * NH * QLEN * DH * 2);
  _Float16* vbf = (_Float16*)alloc((size_t)BSZ * NH * QLEN * DH * 2);
  _Float16* krb = (_Float16*)alloc((size_t)BSZ * NH * RLEN * DH * 2);
  float*    avh = (float*)   alloc((size_t)QLEN * BSZ * DM * 4);
  float*    avg = (float*)   alloc((size_t)QLEN * BSZ * DM * 4);
  float*    poh = (float*)   alloc((size_t)QLEN * BSZ * DM * 4);
  float*    pog = (float*)   alloc((size_t)QLEN * BSZ * DM * 4);

  dim3 blk(128);
  // projections: q_h, q_g (f32 heads), k, v, k_r (f16 heads)
  proj_gemm_f16<0, 0><<<dim3(16, 64),  blk, 0, stream>>>(h, wq, QLEN, nullptr, qh, nullptr);
  proj_gemm_f16<0, 0><<<dim3(16, 64),  blk, 0, stream>>>(g, wq, QLEN, nullptr, qg, nullptr);
  proj_gemm_f16<1, 0><<<dim3(16, 64),  blk, 0, stream>>>(h, wk, QLEN, nullptr, nullptr, kbf);
  proj_gemm_f16<1, 0><<<dim3(16, 64),  blk, 0, stream>>>(h, wv, QLEN, nullptr, nullptr, vbf);
  proj_gemm_f16<1, 0><<<dim3(16, 128), blk, 0, stream>>>(r, wr, RLEN, nullptr, nullptr, krb);

  // attention cores (content + query streams)
  dim3 agrid(QLEN / 16, NH, BSZ);
  attn_core_f16<<<agrid, blk, 0, stream>>>(qh, kbf, vbf, krb, mask_h, segmat,
                                           rwb, rrb, rsb, segemb, avh);
  attn_core_f16<<<agrid, blk, 0, stream>>>(qg, kbf, vbf, krb, mask_g, segmat,
                                           rwb, rrb, rsb, segemb, avg);

  // output projection (wo^T) + residual
  proj_gemm_f16<2, 1><<<dim3(16, 64), blk, 0, stream>>>(avh, wo, QLEN, h, poh, nullptr);
  proj_gemm_f16<2, 1><<<dim3(16, 64), blk, 0, stream>>>(avg, wo, QLEN, g, pog, nullptr);

  // layernorm -> d_out (output_h first, then output_g)
  float* out = (float*)d_out;
  layernorm_k<<<dim3(QLEN * BSZ), dim3(256), 0, stream>>>(poh, gamma, beta, out);
  layernorm_k<<<dim3(QLEN * BSZ), dim3(256), 0, stream>>>(pog, gamma, beta,
                                                          out + (size_t)QLEN * BSZ * DM);
  (void)in_sizes; (void)n_in; (void)out_size; (void)ws_size;
}
